// GraphSem_no_enc_qmix_58334245814475
// MI455X (gfx1250) — matrix operand order repaired and checked by reading, not compiled, not run on hardware
//
#include <hip/hip_runtime.h>
#include <hip/hip_bf16.h>
#include <math.h>

typedef __bf16     v16bf __attribute__((ext_vector_type(16)));
typedef unsigned   uv8   __attribute__((ext_vector_type(8)));
typedef float      v8f   __attribute__((ext_vector_type(8)));

#define NA   8
#define DIM  128
#define EC   64
#define EM   32
#define NW   352   // 256 (W1h) + 32 (Wb1) + 32 (Wf) + 32 (V1w)
#define KEEP 4
#define NT_G 4     // N-tiles of Wg1 (64/16)
#define NT_M 22    // N-tiles of mix weights (352/16)

__device__ inline unsigned short f2bf(float f) {
  unsigned u = __builtin_bit_cast(unsigned, f);
  unsigned r = u + 0x7fffu + ((u >> 16) & 1u);   // round-to-nearest-even
  return (unsigned short)(r >> 16);
}

// Load one pre-swizzled 16-element bf16 fragment (32 bytes) as 2 x b128.
__device__ inline v16bf load_frag16(const void* p) {
  const uint4* q = (const uint4*)p;
  uint4 a = q[0], b = q[1];
  uv8 t;
  t[0] = a.x; t[1] = a.y; t[2] = a.z; t[3] = a.w;
  t[4] = b.x; t[5] = b.y; t[6] = b.z; t[7] = b.w;
  return __builtin_bit_cast(v16bf, t);
}

// ---------------- kernel 0: pack weights to frag-major bf16 ----------------
// B-matrix 32x16 fragment inverse map: e = ((k&16)>>1) + (k&7), lane = (n&15) + 16*((k>>3)&1),
// frag = kc*NT + nt.  Packed index: (frag*32 + lane)*16 + e.
__global__ void k_prep(const float* __restrict__ Wg1, const float* __restrict__ W1h,
                       const float* __restrict__ Wb1, const float* __restrict__ Wf,
                       const float* __restrict__ V1w,
                       unsigned short* __restrict__ wg1p, unsigned short* __restrict__ wcatp) {
  const int TOT1 = DIM * EC, TOT2 = DIM * NW;
  for (int i = blockIdx.x * blockDim.x + threadIdx.x; i < TOT1 + TOT2;
       i += gridDim.x * blockDim.x) {
    if (i < TOT1) {
      int k = i / EC, n = i % EC;
      int dst = (((k >> 5) * NT_G + (n >> 4)) * 32 + ((n & 15) + (((k >> 3) & 1) << 4))) * 16 + (((k & 16) >> 1) + (k & 7));
      wg1p[dst] = f2bf(Wg1[i]);
    } else {
      int j = i - TOT1, k = j / NW, n = j % NW;
      float v;
      if (n < 256)      v = W1h[k * 256 + n];
      else if (n < 288) v = Wb1[k * EM + (n - 256)];
      else if (n < 320) v = Wf [k * EM + (n - 288)];
      else              v = V1w[k * EM + (n - 320)];
      int dst = (((k >> 5) * NT_M + (n >> 4)) * 32 + ((n & 15) + (((k >> 3) & 1) << 4))) * 16 + (((k & 16) >> 1) + (k & 7));
      wcatp[dst] = f2bf(v);
    }
  }
}

// ---------------- kernel 1: adj-normalize + GCN (WMMA) + top-k pool ----------------
__global__ void __launch_bounds__(128) k_gcn_pool(
    const float* __restrict__ obs, const float* __restrict__ adjacency,
    const unsigned short* __restrict__ wg1p,
    const float* __restrict__ bg1, const float* __restrict__ Wg2,
    const float* __restrict__ bg2, unsigned short* __restrict__ poolb) {
  __shared__ __align__(16) unsigned short Wg1s[DIM * EC];   // frag-major, 16 KB
  __shared__ __align__(16) float    xs [4][16 * DIM];       // 32 KB
  __shared__ __align__(16) unsigned axp[4][4 * 32 * 8];     // A frags (bf16 pairs), 16 KB
  __shared__ float hs[4][16 * EC];                          // 16 KB
  __shared__ __align__(16) float adjL[4][16 * NA];
  __shared__ float adjn[4][16 * NA];
  __shared__ float dinvS[4][16];
  __shared__ float scoreS[4][16];
  __shared__ float selv[4][2 * KEEP];
  __shared__ int   seli[4][2 * KEEP];

  const int tid = threadIdx.x, lane = tid & 31, wv = tid >> 5;
  const int g0 = blockIdx.x * 8 + wv * 2;  // this wave: graphs g0, g0+1 (M = 16 rows)

  { // stage packed Wg1 into LDS (whole block)
    const uint4* src = (const uint4*)wg1p;
    uint4* dst = (uint4*)Wg1s;
#pragma unroll
    for (int t = 0; t < 8; ++t) dst[tid + 128 * t] = src[tid + 128 * t];
  }
  { // coalesced float4 loads: 128 adj floats + 2048 obs floats per wave
    const float4* asrc = (const float4*)(adjacency + (size_t)g0 * NA * NA);
    ((float4*)adjL[wv])[lane] = asrc[lane];
    const float4* xsrc = (const float4*)(obs + (size_t)g0 * NA * DIM);
    float4* xdst = (float4*)xs[wv];
#pragma unroll
    for (int t = 0; t < 16; ++t) xdst[lane + 32 * t] = xsrc[lane + 32 * t];
  }
  __syncthreads();

  if (lane < 16) {  // D^-1/2 of (A + I)
    float s = 1.0f;
#pragma unroll
    for (int c = 0; c < NA; ++c) s += adjL[wv][lane * NA + c];
    dinvS[wv][lane] = rsqrtf(s);
  }
  __syncthreads();

#pragma unroll
  for (int t = 0; t < 4; ++t) {  // symmetric renormalization
    int e = lane * 4 + t, m = e >> 3, c = e & 7;
    float a = adjL[wv][e] + (((m & 7) == c) ? 1.0f : 0.0f);
    adjn[wv][e] = a * dinvS[wv][m] * dinvS[wv][(m & 8) + c];
  }
  __syncthreads();

  { // ax = adj_norm @ x (K=8 VALU), written as bf16 pairs in A-fragment order
    int m = lane >> 1, dh = lane & 1, gb = (m & 8);
    float ar[NA];
#pragma unroll
    for (int j = 0; j < NA; ++j) ar[j] = adjn[wv][m * NA + j];
    for (int d = dh * 64; d < dh * 64 + 64; d += 2) {
      float s0 = 0.f, s1 = 0.f;
#pragma unroll
      for (int j = 0; j < NA; ++j) {
        s0 += ar[j] * xs[wv][(gb + j) * DIM + d];
        s1 += ar[j] * xs[wv][(gb + j) * DIM + d + 1];
      }
      int kc = d >> 5, kk = d & 31;
      int h = (kk >> 3) & 1;
      int v = ((kk & 16) >> 2) + ((kk & 7) >> 1);
      axp[wv][(kc * 32 + (m + 16 * h)) * 8 + v] =
          (unsigned)f2bf(s0) | ((unsigned)f2bf(s1) << 16);
    }
  }
  __syncthreads();

  { // h = relu(ax @ Wg1 + bg1) : [16,128]@[128,64] via 16 WMMAs (A frags hoisted)
    const int col = lane & 15, half = lane >> 4;
    v16bf afr[4];
#pragma unroll
    for (int kc = 0; kc < 4; ++kc)
      afr[kc] = load_frag16(&axp[wv][(kc * 32 + lane) * 8]);
#pragma unroll
    for (int nt = 0; nt < NT_G; ++nt) {
      v8f acc = {};
#pragma unroll
      for (int kc = 0; kc < 4; ++kc) {
        v16bf b = load_frag16(Wg1s + ((kc * NT_G + nt) * 32 + lane) * 16);
        acc = __builtin_amdgcn_wmma_f32_16x16x32_bf16(false, afr[kc], false, b,
                                                      (short)0, acc, false, false);
      }
      float bias = bg1[nt * 16 + col];
#pragma unroll
      for (int r = 0; r < 8; ++r)
        hs[wv][(r + half * 8) * EC + nt * 16 + col] = fmaxf(acc[r] + bias, 0.f);
    }
  }
  __syncthreads();

  { // score = tanh((adj @ h) @ Wg2 + bg2)
    int m = lane >> 1, eh = lane & 1, gb = (m & 8);
    float ar[NA];
#pragma unroll
    for (int j = 0; j < NA; ++j) ar[j] = adjn[wv][m * NA + j];
    float p = 0.f;
    for (int e = eh * 32; e < eh * 32 + 32; ++e) {
      float s = 0.f;
#pragma unroll
      for (int j = 0; j < NA; ++j) s += ar[j] * hs[wv][(gb + j) * EC + e];
      p += s * Wg2[e];
    }
    p += __shfl_xor(p, 1, 32);
    if (eh == 0) scoreS[wv][m] = tanhf(p + bg2[0]);
  }
  __syncthreads();

  if (lane < 2) {  // top-4, descending, first-index tie-break (matches jax top_k)
    float sc[NA];
#pragma unroll
    for (int i = 0; i < NA; ++i) sc[i] = scoreS[wv][lane * NA + i];
#pragma unroll
    for (int k = 0; k < KEEP; ++k) {
      int best = 0; float bv = sc[0];
#pragma unroll
      for (int i = 1; i < NA; ++i) if (sc[i] > bv) { bv = sc[i]; best = i; }
      selv[wv][lane * KEEP + k] = bv;
      seli[wv][lane * KEEP + k] = best;
      sc[best] = -__builtin_inff();
    }
  }
  __syncthreads();

  { // pool = mean_k vals[k]*x[idx[k]] -> bf16, written in A-fragment order for k_mix
    int gl = lane >> 4, d0 = (lane & 15) * 8;
    int g = g0 + gl, mtile = g >> 4, mr = g & 15;
    float vv[KEEP]; int ii[KEEP];
#pragma unroll
    for (int k = 0; k < KEEP; ++k) { vv[k] = selv[wv][gl * KEEP + k]; ii[k] = seli[wv][gl * KEEP + k]; }
    unsigned short o[8];
#pragma unroll
    for (int t = 0; t < 8; ++t) {
      int d = d0 + t; float s = 0.f;
#pragma unroll
      for (int k = 0; k < KEEP; ++k) s += vv[k] * xs[wv][(gl * NA + ii[k]) * DIM + d];
      o[t] = f2bf(0.25f * s);
    }
    uint4 pk;
    pk.x = o[0] | (unsigned)o[1] << 16;  pk.y = o[2] | (unsigned)o[3] << 16;
    pk.z = o[4] | (unsigned)o[5] << 16;  pk.w = o[6] | (unsigned)o[7] << 16;
    int kc = d0 >> 5, kk0 = d0 & 31;
    int h = (kk0 >> 3) & 1, e0 = (kk0 & 16) >> 1;          // e0 in {0,8}
    size_t off = ((size_t)(mtile * 4 + kc) * 32 + (mr + 16 * h)) * 16 + e0;
    *(uint4*)(poolb + off) = pk;
  }
}

// ---------------- kernel 2: mix GEMM [16,128]@[128,352] (WMMA) + epilogue ----------------
__global__ void __launch_bounds__(128) k_mix(
    const float* __restrict__ agent_qs, const unsigned short* __restrict__ poolb,
    const unsigned short* __restrict__ wcatp,
    const float* __restrict__ b1h, const float* __restrict__ bb1,
    const float* __restrict__ bfv, const float* __restrict__ V1b,
    const float* __restrict__ V2w, const float* __restrict__ V2b,
    float* __restrict__ out) {
  __shared__ float Z[16 * NW];  // 22 KB

  const int tid = threadIdx.x, lane = tid & 31, wv = tid >> 5;
  const int mtile = blockIdx.x, g0 = mtile * 16;
  const int col = lane & 15, half = lane >> 4;

  // A fragments straight from frag-major pool workspace (L2), reused for all N-tiles
  v16bf afr[4];
#pragma unroll
  for (int kc = 0; kc < 4; ++kc)
    afr[kc] = load_frag16(poolb + ((size_t)(mtile * 4 + kc) * 32 + lane) * 16);

  for (int nt = wv; nt < NT_M; nt += 4) {  // 22 N-tiles split over 4 waves
    v8f acc = {};
#pragma unroll
    for (int kc = 0; kc < 4; ++kc) {
      v16bf b = load_frag16(wcatp + ((size_t)(kc * NT_M + nt) * 32 + lane) * 16);
      acc = __builtin_amdgcn_wmma_f32_16x16x32_bf16(false, afr[kc], false, b,
                                                    (short)0, acc, false, false);
    }
#pragma unroll
    for (int r = 0; r < 8; ++r) Z[(r + half * 8) * NW + nt * 16 + col] = acc[r];
  }
  __syncthreads();

  // epilogue: lane == e (E_MIX == 32 == wave32), 4 graphs per wave
  for (int mm = 0; mm < 4; ++mm) {
    int m = wv * 4 + mm, g = g0 + m;
    float acc = Z[m * NW + 256 + lane] + bb1[lane];  // b1
#pragma unroll
    for (int i = 0; i < NA; ++i)
      acc += agent_qs[(size_t)g * NA + i] * fabsf(Z[m * NW + i * EM + lane] + b1h[i * EM + lane]);
    float hidden = (acc > 0.f) ? acc : expm1f(acc);                       // elu
    float wfv = fabsf(Z[m * NW + 256 + EM + lane] + bfv[lane]);           // |wf|
    float rv  = fmaxf(Z[m * NW + 256 + 2 * EM + lane] + V1b[lane], 0.f) * V2w[lane];
    float part = hidden * wfv + rv;
#pragma unroll
    for (int o = 16; o > 0; o >>= 1) part += __shfl_xor(part, o, 32);
    if (lane == 0) out[g] = part + V2b[0];
  }
}

extern "C" void kernel_launch(void* const* d_in, const int* in_sizes, int n_in,
                              void* d_out, int out_size, void* d_ws, size_t ws_size,
                              hipStream_t stream) {
  const float* agent_qs  = (const float*)d_in[0];
  const float* obs       = (const float*)d_in[1];
  const float* adjacency = (const float*)d_in[2];
  const float* Wg1 = (const float*)d_in[3];
  const float* bg1 = (const float*)d_in[4];
  const float* Wg2 = (const float*)d_in[5];
  const float* bg2 = (const float*)d_in[6];
  const float* W1h = (const float*)d_in[7];
  const float* b1h = (const float*)d_in[8];
  const float* Wb1 = (const float*)d_in[9];
  const float* bb1 = (const float*)d_in[10];
  const float* Wf  = (const float*)d_in[11];
  const float* bf_ = (const float*)d_in[12];
  const float* V1w = (const float*)d_in[13];
  const float* V1b = (const float*)d_in[14];
  const float* V2w = (const float*)d_in[15];
  const float* V2b = (const float*)d_in[16];
  float* out = (float*)d_out;

  const int G = in_sizes[0] / NA;  // 32768
  unsigned short* wg1p  = (unsigned short*)d_ws;
  unsigned short* wcatp = wg1p + DIM * EC;
  unsigned short* poolb = wcatp + DIM * NW;  // G*DIM bf16 (~8 MB, L2-resident)

  k_prep<<<64, 256, 0, stream>>>(Wg1, W1h, Wb1, Wf, V1w, wg1p, wcatp);
  k_gcn_pool<<<G / 8, 128, 0, stream>>>(obs, adjacency, wg1p, bg1, Wg2, bg2, poolb);
  k_mix<<<G / 16, 128, 0, stream>>>(agent_qs, poolb, wcatp, b1h, bb1, bf_, V1b, V2w, V2b, out);
}
